// Matching_29686813950657
// MI455X (gfx1250) — matrix operand order repaired and checked by reading, not compiled, not run on hardware
//
#include <hip/hip_runtime.h>
#include <hip/hip_bf16.h>

// ---------------- CDNA5 WMMA types & helpers ----------------
typedef __attribute__((ext_vector_type(16))) __bf16 v16bf;
typedef __attribute__((ext_vector_type(8)))  __bf16 v8bf;
typedef __attribute__((ext_vector_type(8)))  float  v8f;

#define TT   384         // TX == TY == 384
#define BB   48
#define DD   512         // D_IN
#define HH   256
#define G4   1024        // 4*H
#define MM   (TT*BB)     // 18432 rows for (t,b)-flattened GEMMs

// native f32<->bf16 converts (v_cvt_*_bf16 on gfx1250, RNE)
__device__ __forceinline__ __bf16 f2bf(float f) { return (__bf16)f; }
__device__ __forceinline__ unsigned short f2bfu(float f) {
    return __builtin_bit_cast(unsigned short, (__bf16)f);
}
__device__ __forceinline__ float bf2f(__bf16 h) { return (float)h; }
__device__ __forceinline__ float sigf(float x) { return 1.f / (1.f + __expf(-x)); }

__device__ __forceinline__ v8f wmma_bf(v16bf a, v16bf b, v8f c) {
    // emits v_wmma_f32_16x16x32_bf16
    return __builtin_amdgcn_wmma_f32_16x16x32_bf16(false, a, false, b, (short)0, c, false, false);
}

// A operand (16x32 bf16): lane row = l&15; 16 contiguous K [kg, kg+16), kg=(l>>4)*16
__device__ __forceinline__ v16bf load_a_f32(const float* row, int k0, int l) {
    const int kg = (l >> 4) << 4;
    const float4* p0 = (const float4*)(row + k0 + kg);
    const float4* p1 = (const float4*)(row + k0 + 8 + kg);
    float4 a0 = p0[0], a1 = p0[1], b0 = p1[0], b1 = p1[1];
    v16bf v;
    v[0]=f2bf(a0.x); v[1]=f2bf(a0.y); v[2]=f2bf(a0.z); v[3]=f2bf(a0.w);
    v[4]=f2bf(a1.x); v[5]=f2bf(a1.y); v[6]=f2bf(a1.z); v[7]=f2bf(a1.w);
    v[8]=f2bf(b0.x); v[9]=f2bf(b0.y); v[10]=f2bf(b0.z); v[11]=f2bf(b0.w);
    v[12]=f2bf(b1.x); v[13]=f2bf(b1.y); v[14]=f2bf(b1.z); v[15]=f2bf(b1.w);
    return v;
}
__device__ __forceinline__ v16bf load_a_bf16(const __bf16* row, int k0, int l) {
    const int kg = (l >> 4) << 4;
    v8bf c0 = *(const v8bf*)(row + k0 + kg);
    v8bf c1 = *(const v8bf*)(row + k0 + 8 + kg);
    v16bf v;
#pragma unroll
    for (int i = 0; i < 8; ++i) { v[i] = c0[i]; v[8 + i] = c1[i]; }
    return v;
}
// B operand (32x16): lane col = l&15, 16 contiguous K at (l>>4)*16 (caller applies)
__device__ __forceinline__ v16bf load_b_f32(const float* p) {
    v16bf v;
#pragma unroll
    for (int c = 0; c < 4; ++c) {
        float4 q = ((const float4*)p)[c];
        v[c*4+0]=f2bf(q.x); v[c*4+1]=f2bf(q.y); v[c*4+2]=f2bf(q.z); v[c*4+3]=f2bf(q.w);
    }
    return v;
}
__device__ __forceinline__ v16bf load_b_bf16(const __bf16* p) {
    v8bf c0 = *(const v8bf*)(p);
    v8bf c1 = *(const v8bf*)(p + 8);
    v16bf v;
#pragma unroll
    for (int i = 0; i < 8; ++i) { v[i] = c0[i]; v[8 + i] = c1[i]; }
    return v;
}

// ---------------- K1: attention logits  W[b][x][y] = x[x,b,:] . y[y,b,:] ----------------
__global__ void logits_kernel(const float* __restrict__ X, const float* __restrict__ Y,
                              float* __restrict__ W) {
    const int l = threadIdx.x;
    const int x0 = blockIdx.x * 16, y0 = blockIdx.y * 16, b = blockIdx.z;
    const float* arow = X + (size_t)(x0 + (l & 15)) * (BB * DD) + (size_t)b * DD;
    const float* bcol = Y + (size_t)(y0 + (l & 15)) * (BB * DD) + (size_t)b * DD;
    v8f acc = {};
    for (int k0 = 0; k0 < DD; k0 += 32) {
        v16bf a = load_a_f32(arow, k0, l);
        v16bf bm = load_b_f32(bcol + k0 + ((l >> 4) << 4));
        acc = wmma_bf(a, bm, acc);
    }
    const int yy = y0 + (l & 15), xb = x0 + ((l >> 4) << 3);
#pragma unroll
    for (int r = 0; r < 8; ++r)
        W[(size_t)b * (TT * TT) + (size_t)(xb + r) * TT + yy] = acc[r];
}

// ---------------- K2: row stats over y (for beta) ----------------
__global__ void row_stats_kernel(const float* __restrict__ W, const float* __restrict__ ymask,
                                 float* __restrict__ rmax, float* __restrict__ rsum) {
    const int wid = blockIdx.x * 8 + (threadIdx.x >> 5);
    const int l = threadIdx.x & 31;
    const int b = wid / TT, xi = wid % TT;
    const float* row = W + (size_t)b * (TT * TT) + (size_t)xi * TT;
    float v[12];
    float mx = -3.4e38f;
#pragma unroll
    for (int i = 0; i < 12; ++i) { v[i] = row[l + 32 * i]; mx = fmaxf(mx, v[i]); }
#pragma unroll
    for (int o = 16; o >= 1; o >>= 1) mx = fmaxf(mx, __shfl_xor(mx, o, 32));
    float s = 0.f;
#pragma unroll
    for (int i = 0; i < 12; ++i) { int yy = l + 32 * i; s += __expf(v[i] - mx) * ymask[yy * BB + b]; }
#pragma unroll
    for (int o = 16; o >= 1; o >>= 1) s += __shfl_xor(s, o, 32);
    if (l == 0) { rmax[wid] = mx; rsum[wid] = s; }
}

// ---------------- K3: col stats over x (for alpha) ----------------
__global__ void col_stats_kernel(const float* __restrict__ W, const float* __restrict__ xmask,
                                 float* __restrict__ cmax, float* __restrict__ csum) {
    const int wid = blockIdx.x * 8 + (threadIdx.x >> 5);
    const int l = threadIdx.x & 31;
    const int b = wid / TT, yi = wid % TT;
    const float* col = W + (size_t)b * (TT * TT) + yi;
    float v[12];
    float mx = -3.4e38f;
#pragma unroll
    for (int i = 0; i < 12; ++i) { v[i] = col[(size_t)(l + 32 * i) * TT]; mx = fmaxf(mx, v[i]); }
#pragma unroll
    for (int o = 16; o >= 1; o >>= 1) mx = fmaxf(mx, __shfl_xor(mx, o, 32));
    float s = 0.f;
#pragma unroll
    for (int i = 0; i < 12; ++i) { int xx = l + 32 * i; s += __expf(v[i] - mx) * xmask[xx * BB + b]; }
#pragma unroll
    for (int o = 16; o >= 1; o >>= 1) s += __shfl_xor(s, o, 32);
    if (l == 0) { cmax[wid] = mx; csum[wid] = s; }
}

// ---------------- K4: materialize alphaT[b][y][x] and beta[b][x][y] (bf16) ----------------
__global__ void softmax_kernel(const float* __restrict__ W, const float* __restrict__ xmask,
                               const float* __restrict__ ymask,
                               const float* __restrict__ rmax, const float* __restrict__ rsum,
                               const float* __restrict__ cmax, const float* __restrict__ csum,
                               unsigned short* __restrict__ alphaT, unsigned short* __restrict__ betaS) {
    const int idx = blockIdx.x * 256 + threadIdx.x;        // < 48*384*384
    const int b = idx / (TT * TT);
    const int rem = idx % (TT * TT);
    const int xi = rem / TT, yi = rem % TT;
    const float w = W[idx];
    const float al = __expf(w - cmax[b * TT + yi]) * xmask[xi * BB + b] / csum[b * TT + yi];
    const float be = __expf(w - rmax[b * TT + xi]) * ymask[yi * BB + b] / rsum[b * TT + xi];
    alphaT[(size_t)b * (TT * TT) + (size_t)yi * TT + xi] = f2bfu(al);
    betaS[idx] = f2bfu(be);
}

// ---------------- K5: attention GEMM  C(T,B,D) = A_bf16[b] (384x384) * src[:,b,:] ----------------
__global__ void att_gemm_kernel(const unsigned short* __restrict__ Abf16,
                                const float* __restrict__ Bsrc, float* __restrict__ Cout) {
    __shared__ float bt[32 * 20];       // 32 K-rows x 16 N cols, padded stride 20
    const int l = threadIdx.x;
    const int m0 = blockIdx.x * 16, n0 = blockIdx.y * 16, b = blockIdx.z;
    const __bf16* arow = (const __bf16*)Abf16 + (size_t)b * (TT * TT) + (size_t)(m0 + (l & 15)) * TT;
    v8f acc = {};
    for (int k0 = 0; k0 < TT; k0 += 32) {
        const float* brow = Bsrc + (size_t)(k0 + l) * (BB * DD) + (size_t)b * DD + n0;
        float4 b0 = ((const float4*)brow)[0];
        float4 b1 = ((const float4*)brow)[1];
        float4 b2 = ((const float4*)brow)[2];
        float4 b3 = ((const float4*)brow)[3];
        __syncthreads();
        ((float4*)(bt + l * 20))[0] = b0;
        ((float4*)(bt + l * 20))[1] = b1;
        ((float4*)(bt + l * 20))[2] = b2;
        ((float4*)(bt + l * 20))[3] = b3;
        __syncthreads();
        v16bf a = load_a_bf16(arow, k0, l);
        v16bf bm;
        const int kb = (l >> 4) << 4, nn = l & 15;
#pragma unroll
        for (int j = 0; j < 16; ++j) bm[j] = f2bf(bt[(kb + j) * 20 + nn]);
        acc = wmma_bf(a, bm, acc);
    }
    const int nn = n0 + (l & 15), mb = m0 + ((l >> 4) << 3);
#pragma unroll
    for (int r = 0; r < 8; ++r)
        Cout[(size_t)(mb + r) * (BB * DD) + (size_t)b * DD + nn] = acc[r];
}

// ---------------- K6: convert Whh (fwd|bwd) to bf16 ----------------
__global__ void cvt_whh_kernel(const float* __restrict__ wf, const float* __restrict__ wb,
                               unsigned short* __restrict__ outp) {
    const int i = blockIdx.x * 256 + threadIdx.x;          // < 2*1024*256
    const int n = G4 * HH;
    outp[i] = (i < n) ? f2bfu(wf[i]) : f2bfu(wb[i - n]);
}

// ---------------- K7: MLP over on-the-fly concat [x, a, x-a, x*a], relu ----------------
__global__ void mlp_kernel(const float* __restrict__ src, const float* __restrict__ att,
                           const float* __restrict__ Wm, const float* __restrict__ bm,
                           float* __restrict__ out) {
    const int l = threadIdx.x;
    const int m0 = blockIdx.x * 16, n0 = blockIdx.y * 16;
    const float* rowx = src + (size_t)(m0 + (l & 15)) * DD;
    const float* rowa = att + (size_t)(m0 + (l & 15)) * DD;
    const float* wrow = Wm + (size_t)(n0 + (l & 15)) * (4 * DD);
    v8f acc = {};
    for (int k0 = 0; k0 < 4 * DD; k0 += 32) {
        const int seg = k0 >> 9;
        const int kk = (k0 & (DD - 1)) + ((l >> 4) << 4);
        v16bf a;
#pragma unroll
        for (int c = 0; c < 2; ++c) {
            const float4* px = (const float4*)(rowx + kk + c * 8);
            const float4* pa = (const float4*)(rowa + kk + c * 8);
            float4 x0 = px[0], x1 = px[1], a0 = pa[0], a1 = pa[1];
            float xs[8] = {x0.x, x0.y, x0.z, x0.w, x1.x, x1.y, x1.z, x1.w};
            float as[8] = {a0.x, a0.y, a0.z, a0.w, a1.x, a1.y, a1.z, a1.w};
#pragma unroll
            for (int j = 0; j < 8; ++j) {
                float xv = xs[j], av = as[j];
                float v = (seg == 0) ? xv : (seg == 1) ? av : (seg == 2) ? (xv - av) : (xv * av);
                a[c * 8 + j] = f2bf(v);
            }
        }
        v16bf bmv = load_b_f32(wrow + k0 + ((l >> 4) << 4));
        acc = wmma_bf(a, bmv, acc);
    }
    const int h = n0 + (l & 15), mb = m0 + ((l >> 4) << 3);
    const float bias = bm[h];
#pragma unroll
    for (int r = 0; r < 8; ++r) {
        float v = acc[r] + bias;
        out[(size_t)(mb + r) * HH + h] = v > 0.f ? v : 0.f;
    }
}

// ---------------- K8: xw = mlp @ Wih^T + (bih + bhh) ----------------
__global__ void xw_kernel(const float* __restrict__ mlp, const float* __restrict__ Wih,
                          const float* __restrict__ bih, const float* __restrict__ bhh,
                          float* __restrict__ xwOut) {
    const int l = threadIdx.x;
    const int m0 = blockIdx.x * 16, n0 = blockIdx.y * 16;
    const float* arow = mlp + (size_t)(m0 + (l & 15)) * HH;
    const float* wrow = Wih + (size_t)(n0 + (l & 15)) * HH;
    v8f acc = {};
    for (int k0 = 0; k0 < HH; k0 += 32) {
        v16bf a = load_a_f32(arow, k0, l);
        v16bf b = load_b_f32(wrow + k0 + ((l >> 4) << 4));
        acc = wmma_bf(a, b, acc);
    }
    const int n = n0 + (l & 15), mb = m0 + ((l >> 4) << 3);
    const float bias = bih[n] + bhh[n];
#pragma unroll
    for (int r = 0; r < 8; ++r)
        xwOut[(size_t)(mb + r) * G4 + n] = acc[r] + bias;
}

// ---------------- K9: persistent LSTM recurrence (one WG per input x direction) ----------------
__global__ __launch_bounds__(768) void lstm_kernel(const float* __restrict__ xwAll,
                                                   const unsigned short* __restrict__ whhBf,
                                                   const float* __restrict__ xmask,
                                                   const float* __restrict__ ymask,
                                                   float* __restrict__ outx, float* __restrict__ outy) {
    __shared__ __bf16 hS[BB * HH];   // h in A-operand-ready bf16
    __shared__ float  cS[BB * HH];
    const int unit = blockIdx.x;               // 0..3 = (input<<1)|dir
    const int input = unit >> 1, dir = unit & 1;
    const float* xw = xwAll + (size_t)unit * ((size_t)MM * G4);
    const __bf16* whh = (const __bf16*)whhBf + (size_t)dir * (G4 * HH);
    const float* mask = input ? ymask : xmask;
    float* out = input ? outy : outx;
    const int colOff = dir * HH;
    const int tid = threadIdx.x, l = tid & 31, w = tid >> 5;

    for (int i = tid; i < BB * HH; i += 768) { hS[i] = f2bf(0.f); cS[i] = 0.f; }
    __syncthreads();

    int jt[2], mt[2];
#pragma unroll
    for (int ji = 0; ji < 2; ++ji) { int job = w * 2 + ji; jt[ji] = job / 3; mt[ji] = job % 3; }

    for (int s = 0; s < TT; ++s) {
        const int t = dir ? (TT - 1 - s) : s;
        // prefetch next timestep's xw gate rows for this wave's tiles (global_prefetch_b8)
        if (s + 1 < TT) {
            const int tn = dir ? (t - 1) : (t + 1);
#pragma unroll
            for (int ji = 0; ji < 2; ++ji) {
                const int m = mt[ji] * 16 + ((l >> 4) << 3);
                const int n = jt[ji] * 16 + (l & 15);
                __builtin_prefetch(xw + ((size_t)tn * BB + m) * G4 + n, 0, 1);
            }
        }
        v8f acc[2][4];
#pragma unroll
        for (int ji = 0; ji < 2; ++ji)
#pragma unroll
            for (int g = 0; g < 4; ++g) { v8f z = {}; acc[ji][g] = z; }

#pragma unroll
        for (int ji = 0; ji < 2; ++ji) {
            const int m0 = mt[ji] * 16, n0 = jt[ji] * 16;
            const __bf16* hrow = hS + (size_t)(m0 + (l & 15)) * HH;
            for (int k0 = 0; k0 < HH; k0 += 32) {
                v16bf a = load_a_bf16(hrow, k0, l);
#pragma unroll
                for (int g = 0; g < 4; ++g) {
                    const __bf16* bp = whh + (size_t)(g * HH + n0 + (l & 15)) * HH + k0 + ((l >> 4) << 4);
                    v16bf bm = load_b_bf16(bp);
                    acc[ji][g] = wmma_bf(a, bm, acc[ji][g]);
                }
            }
        }
        __syncthreads();   // all reads of hS done before anyone rewrites it

#pragma unroll
        for (int ji = 0; ji < 2; ++ji) {
            const int m0 = mt[ji] * 16, n0 = jt[ji] * 16;
            const int n = n0 + (l & 15);
#pragma unroll
            for (int r = 0; r < 8; ++r) {
                const int m = m0 + ((l >> 4) << 3) + r;
                const size_t xrow = ((size_t)t * BB + m) * G4;
                float iv = sigf(acc[ji][0][r] + xw[xrow + n]);
                float fv = sigf(acc[ji][1][r] + xw[xrow + HH + n]);
                float gv = tanhf(acc[ji][2][r] + xw[xrow + 2 * HH + n]);
                float ov = sigf(acc[ji][3][r] + xw[xrow + 3 * HH + n]);
                const int hi = m * HH + n;
                float cold = cS[hi];
                float cnew = fv * cold + iv * gv;
                float hnew = ov * tanhf(cnew);
                float mv = mask[t * BB + m];
                float hold = bf2f(hS[hi]);
                cS[hi] = mv > 0.f ? cnew : cold;
                hS[hi] = f2bf(mv > 0.f ? hnew : hold);
                out[((size_t)t * BB + m) * (2 * HH) + colOff + n] = hnew * mv;
            }
        }
        __syncthreads();   // h updated before next step's GEMM
    }
}

// ---------------- launch ----------------
extern "C" void kernel_launch(void* const* d_in, const int* in_sizes, int n_in,
                              void* d_out, int out_size, void* d_ws, size_t ws_size,
                              hipStream_t stream) {
    (void)in_sizes; (void)n_in; (void)out_size; (void)ws_size;
    const float* inp_x = (const float*)d_in[0];
    const float* xmask = (const float*)d_in[2];
    const float* inp_y = (const float*)d_in[3];
    const float* ymask = (const float*)d_in[5];
    const float* Wmlp  = (const float*)d_in[6];
    const float* bmlp  = (const float*)d_in[7];
    const float* Wih_f = (const float*)d_in[8];
    const float* Whh_f = (const float*)d_in[9];
    const float* bih_f = (const float*)d_in[10];
    const float* bhh_f = (const float*)d_in[11];
    const float* Wih_b = (const float*)d_in[12];
    const float* Whh_b = (const float*)d_in[13];
    const float* bih_b = (const float*)d_in[14];
    const float* bhh_b = (const float*)d_in[15];

    float* outx = (float*)d_out;
    float* outy = outx + (size_t)TT * BB * (2 * HH);

    size_t off = 0;
    auto alloc = [&](size_t bytes) -> void* {
        void* p = (char*)d_ws + off;
        off += (bytes + 255) & ~(size_t)255;
        return p;
    };
    float* W              = (float*)alloc((size_t)BB * TT * TT * 4);
    float* rmax           = (float*)alloc((size_t)BB * TT * 4);
    float* rsum           = (float*)alloc((size_t)BB * TT * 4);
    float* cmax           = (float*)alloc((size_t)BB * TT * 4);
    float* csum           = (float*)alloc((size_t)BB * TT * 4);
    unsigned short* alphaT = (unsigned short*)alloc((size_t)BB * TT * TT * 2);
    unsigned short* betaS  = (unsigned short*)alloc((size_t)BB * TT * TT * 2);
    float* xatt           = (float*)alloc((size_t)MM * DD * 4);
    float* yatt           = (float*)alloc((size_t)MM * DD * 4);
    float* mlpx           = (float*)alloc((size_t)MM * HH * 4);
    float* mlpy           = (float*)alloc((size_t)MM * HH * 4);
    unsigned short* whhBf = (unsigned short*)alloc((size_t)2 * G4 * HH * 2);
    float* xwAll          = (float*)alloc((size_t)4 * MM * G4 * 4);

    logits_kernel<<<dim3(TT / 16, TT / 16, BB), 32, 0, stream>>>(inp_x, inp_y, W);
    row_stats_kernel<<<(BB * TT) / 8, 256, 0, stream>>>(W, ymask, rmax, rsum);
    col_stats_kernel<<<(BB * TT) / 8, 256, 0, stream>>>(W, xmask, cmax, csum);
    softmax_kernel<<<(BB * TT * TT) / 256, 256, 0, stream>>>(W, xmask, ymask, rmax, rsum,
                                                             cmax, csum, alphaT, betaS);
    att_gemm_kernel<<<dim3(TT / 16, DD / 16, BB), 32, 0, stream>>>(alphaT, inp_x, yatt);
    att_gemm_kernel<<<dim3(TT / 16, DD / 16, BB), 32, 0, stream>>>(betaS, inp_y, xatt);
    mlp_kernel<<<dim3(MM / 16, HH / 16), 32, 0, stream>>>(inp_x, xatt, Wmlp, bmlp, mlpx);
    mlp_kernel<<<dim3(MM / 16, HH / 16), 32, 0, stream>>>(inp_y, yatt, Wmlp, bmlp, mlpy);
    cvt_whh_kernel<<<(2 * G4 * HH) / 256, 256, 0, stream>>>(Whh_f, Whh_b, whhBf);

    const size_t XWU = (size_t)MM * G4;
    xw_kernel<<<dim3(MM / 16, G4 / 16), 32, 0, stream>>>(mlpx, Wih_f, bih_f, bhh_f, xwAll + 0 * XWU);
    xw_kernel<<<dim3(MM / 16, G4 / 16), 32, 0, stream>>>(mlpx, Wih_b, bih_b, bhh_b, xwAll + 1 * XWU);
    xw_kernel<<<dim3(MM / 16, G4 / 16), 32, 0, stream>>>(mlpy, Wih_f, bih_f, bhh_f, xwAll + 2 * XWU);
    xw_kernel<<<dim3(MM / 16, G4 / 16), 32, 0, stream>>>(mlpy, Wih_b, bih_b, bhh_b, xwAll + 3 * XWU);

    lstm_kernel<<<4, 768, 0, stream>>>(xwAll, whhBf, xmask, ymask, outx, outy);
}
